// GNN_3332894622041
// MI455X (gfx1250) — matrix operand order repaired and checked by reading, not compiled, not run on hardware
//
#include <hip/hip_runtime.h>
#include <hip/hip_bf16.h>
#include <stdint.h>

#define N_NODES 50000
#define N_EDGES 800000
#define FDIM    128
#define NEG_SLOPE 0.2f

typedef __attribute__((ext_vector_type(16))) __bf16 v16bf;
typedef __attribute__((ext_vector_type(8)))  float  v8f;

union Frag {
    uint4        u4[2];
    unsigned int u[8];
    v16bf        bf;
};

// ---------- helpers ----------
__device__ __forceinline__ unsigned short f2bf(float f) {
    unsigned u = __float_as_uint(f);
    u += 0x7FFFu + ((u >> 16) & 1u);   // round-to-nearest-even
    return (unsigned short)(u >> 16);
}
__device__ __forceinline__ unsigned orderF(float f) {
    unsigned u = __float_as_uint(f);
    return (u & 0x80000000u) ? ~u : (u | 0x80000000u);
}
__device__ __forceinline__ float unorderF(unsigned o) {
    unsigned u = (o & 0x80000000u) ? (o & 0x7FFFFFFFu) : ~o;
    return __uint_as_float(u);
}
__device__ __forceinline__ float lrelu(float v) {
    return v > 0.f ? v : NEG_SLOPE * v;
}

// ---------- fp32 -> bf16 (optional fused ReLU) ----------
__global__ void k_to_bf16(const float* __restrict__ in,
                          unsigned short* __restrict__ out,
                          int n, int do_relu) {
    int i = blockIdx.x * blockDim.x + threadIdx.x;
    if (i >= n) return;
    float v = in[i];
    if (do_relu) v = v > 0.f ? v : 0.f;
    out[i] = f2bf(v);
}

// ---------- pack weight [128 K x 128 N] into WMMA B-fragment order ----------
// out index = ((n_tile*4 + kc)*32 + lane)*8 + v  ; each uint = 2 packed bf16
__global__ void k_pack_w(const float* __restrict__ W, unsigned int* __restrict__ out) {
    int t = blockIdx.x * blockDim.x + threadIdx.x;
    if (t >= 8192) return;
    int n_tile = t >> 10;
    int kc     = (t >> 8) & 3;
    int lane   = (t >> 3) & 31;
    int v      = t & 7;
    int col = n_tile * 16 + (lane & 15);
    int k0  = kc * 32 + ((lane >> 4) * 8) + 2 * v + ((v >> 2) * 8);
    unsigned lo = f2bf(W[k0 * FDIM + col]);
    unsigned hi = f2bf(W[(k0 + 1) * FDIM + col]);
    out[t] = lo | (hi << 16);
}

// ---------- WMMA GEMM: out[N x 128] = Abf[N x 128] @ Bpack + bias ----------
// block = 256 threads = 8 waves; wave w owns the 16x16 tile at cols [16w,16w+16)
// grid.x = N/16 row tiles; K = 128 -> 4 x v_wmma_f32_16x16x32_bf16 per tile
__global__ void __launch_bounds__(256)
k_gemm_wmma(const unsigned short* __restrict__ Abf,
            const unsigned int*  __restrict__ Bpack,
            const float*         __restrict__ bias,
            float*               __restrict__ out) {
    int m0    = blockIdx.x * 16;
    int wave  = threadIdx.x >> 5;
    int lane  = threadIdx.x & 31;
    int laneL = lane & 15;
    int laneH = lane >> 4;
    int n0    = wave * 16;

    v8f acc;
    float bv = bias[n0 + laneL];
#pragma unroll
    for (int j = 0; j < 8; ++j) acc[j] = bv;

    // A row for this lane: 128 bf16 = 16 x uint4 (8 bf16 per uint4)
    const uint4* Arow = (const uint4*)(Abf + (size_t)(m0 + laneL) * FDIM);

#pragma unroll
    for (int kc = 0; kc < 4; ++kc) {
        Frag a, b;
        // A 16x32 bf16 layout: v0..3 = K[kc*32 + laneH*8 .. +7], v4..7 = +16
        a.u4[0] = Arow[kc * 4 + laneH];
        a.u4[1] = Arow[kc * 4 + 2 + laneH];
        const uint4* bp = (const uint4*)(Bpack + (size_t)((wave * 4 + kc) * 32 + lane) * 8);
        b.u4[0] = bp[0];
        b.u4[1] = bp[1];
        acc = __builtin_amdgcn_wmma_f32_16x16x32_bf16(
            false, a.bf, false, b.bf, (short)0, acc, false, false);
    }

#pragma unroll
    for (int j = 0; j < 8; ++j) {
        int row = m0 + laneH * 8 + j;          // C layout: vgpr j, lanes 0-15 -> M=j, 16-31 -> M=j+8
        out[(size_t)row * FDIM + n0 + laneL] = acc[j];
    }
}

// ---------- per-node init: agg = bias (broadcast), smax/denom reset ----------
__global__ void k_init_nodes(float* __restrict__ agg, const float* __restrict__ bias,
                             unsigned int* __restrict__ smax, float* __restrict__ denom) {
    int i = blockIdx.x * blockDim.x + threadIdx.x;
    if (i < N_NODES * FDIM) agg[i] = bias[i & (FDIM - 1)];
    if (i < N_NODES) { smax[i] = 0u; denom[i] = 0.f; }
}

// ---------- pass 1: wave-per-edge score + segment max ----------
__global__ void __launch_bounds__(256)
k_edge_score(const float* __restrict__ xl, const float* __restrict__ xr,
             const float* __restrict__ att,
             const int* __restrict__ src, const int* __restrict__ dst,
             float* __restrict__ s_out, unsigned int* __restrict__ smax) {
    int e    = (blockIdx.x * blockDim.x + threadIdx.x) >> 5;
    int lane = threadIdx.x & 31;
    if (e >= N_EDGES) return;
    int sn = src[e], dn = dst[e];
    float4 a = ((const float4*)(xl + (size_t)sn * FDIM))[lane];
    float4 b = ((const float4*)(xr + (size_t)dn * FDIM))[lane];
    float4 w = ((const float4*)att)[lane];
    float sum = lrelu(a.x + b.x) * w.x + lrelu(a.y + b.y) * w.y +
                lrelu(a.z + b.z) * w.z + lrelu(a.w + b.w) * w.w;
#pragma unroll
    for (int off = 16; off; off >>= 1) sum += __shfl_xor(sum, off, 32);
    if (lane == 0) {
        s_out[e] = sum;
        atomicMax(&smax[dn], orderF(sum));
    }
}

// ---------- pass 2: thread-per-edge exp + denom ----------
__global__ void k_edge_exp(const float* __restrict__ s_in, const int* __restrict__ dst,
                           const unsigned int* __restrict__ smax,
                           float* __restrict__ p_out, float* __restrict__ denom) {
    int e = blockIdx.x * blockDim.x + threadIdx.x;
    if (e >= N_EDGES) return;
    int dn = dst[e];
    float p = __expf(s_in[e] - unorderF(smax[dn]));
    p_out[e] = p;
    atomicAdd(&denom[dn], p);
}

// ---------- pass 3: wave-per-edge weighted scatter-add ----------
__global__ void __launch_bounds__(256)
k_edge_agg(const float* __restrict__ xl,
           const int* __restrict__ src, const int* __restrict__ dst,
           const float* __restrict__ p, const float* __restrict__ denom,
           float* __restrict__ out) {
    int e    = (blockIdx.x * blockDim.x + threadIdx.x) >> 5;
    int lane = threadIdx.x & 31;
    if (e >= N_EDGES) return;
    int sn = src[e], dn = dst[e];
    float alpha = p[e] / denom[dn];
    float4 m = ((const float4*)(xl + (size_t)sn * FDIM))[lane];
    float* o = out + (size_t)dn * FDIM + lane * 4;
    atomicAdd(o + 0, alpha * m.x);
    atomicAdd(o + 1, alpha * m.y);
    atomicAdd(o + 2, alpha * m.z);
    atomicAdd(o + 3, alpha * m.w);
}

// ===================== host side =====================
static inline size_t alignup(size_t v) { return (v + 255) & ~(size_t)255; }

extern "C" void kernel_launch(void* const* d_in, const int* in_sizes, int n_in,
                              void* d_out, int out_size, void* d_ws, size_t ws_size,
                              hipStream_t stream) {
    (void)in_sizes; (void)n_in; (void)out_size; (void)ws_size;

    const float* x    = (const float*)d_in[0];
    const int*   eidx = (const int*)d_in[1];
    const float* Wl1  = (const float*)d_in[2];
    const float* bl1  = (const float*)d_in[3];
    const float* Wr1  = (const float*)d_in[4];
    const float* br1  = (const float*)d_in[5];
    const float* att1 = (const float*)d_in[6];
    const float* b1   = (const float*)d_in[7];
    const float* Wl2  = (const float*)d_in[8];
    const float* bl2  = (const float*)d_in[9];
    const float* Wr2  = (const float*)d_in[10];
    const float* br2  = (const float*)d_in[11];
    const float* att2 = (const float*)d_in[12];
    const float* b2   = (const float*)d_in[13];
    float* out = (float*)d_out;

    const int* src = eidx;            // edge_idx[0]
    const int* dst = eidx + N_EDGES;  // edge_idx[1]

    // workspace layout
    char* w = (char*)d_ws;
    auto alloc = [&](size_t bytes) -> void* { void* p = (void*)w; w += alignup(bytes); return p; };
    unsigned short* xbf   = (unsigned short*)alloc((size_t)N_NODES * FDIM * 2);
    unsigned short* hbf   = (unsigned short*)alloc((size_t)N_NODES * FDIM * 2);
    unsigned int*   pWl1  = (unsigned int*)alloc(8192 * 4);
    unsigned int*   pWr1  = (unsigned int*)alloc(8192 * 4);
    unsigned int*   pWl2  = (unsigned int*)alloc(8192 * 4);
    unsigned int*   pWr2  = (unsigned int*)alloc(8192 * 4);
    float*          xlb   = (float*)alloc((size_t)N_NODES * FDIM * 4);
    float*          xrb   = (float*)alloc((size_t)N_NODES * FDIM * 4);
    float*          agg1  = (float*)alloc((size_t)N_NODES * FDIM * 4);
    float*          sbuf  = (float*)alloc((size_t)N_EDGES * 4);
    float*          pbuf  = (float*)alloc((size_t)N_EDGES * 4);
    unsigned int*   smax  = (unsigned int*)alloc((size_t)N_NODES * 4);
    float*          denom = (float*)alloc((size_t)N_NODES * 4);

    const int TB = 256;
    const int gElem  = (N_NODES * FDIM + TB - 1) / TB;  // element-wise over N*128
    const int gGemm  = N_NODES / 16;                    // 3125 row tiles
    const int gWaveE = (N_EDGES * 32 + TB - 1) / TB;    // wave-per-edge
    const int gEdge  = (N_EDGES + TB - 1) / TB;

    // ---- layer 1 ----
    k_to_bf16<<<gElem, TB, 0, stream>>>(x, xbf, N_NODES * FDIM, 0);
    k_pack_w<<<32, TB, 0, stream>>>(Wl1, pWl1);
    k_pack_w<<<32, TB, 0, stream>>>(Wr1, pWr1);
    k_pack_w<<<32, TB, 0, stream>>>(Wl2, pWl2);
    k_pack_w<<<32, TB, 0, stream>>>(Wr2, pWr2);
    k_gemm_wmma<<<gGemm, TB, 0, stream>>>(xbf, pWl1, bl1, xlb);
    k_gemm_wmma<<<gGemm, TB, 0, stream>>>(xbf, pWr1, br1, xrb);
    k_init_nodes<<<gElem, TB, 0, stream>>>(agg1, b1, smax, denom);
    k_edge_score<<<gWaveE, TB, 0, stream>>>(xlb, xrb, att1, src, dst, sbuf, smax);
    k_edge_exp<<<gEdge, TB, 0, stream>>>(sbuf, dst, smax, pbuf, denom);
    k_edge_agg<<<gWaveE, TB, 0, stream>>>(xlb, src, dst, pbuf, denom, agg1);

    // ---- layer 2 (ReLU fused into bf16 convert) ----
    k_to_bf16<<<gElem, TB, 0, stream>>>(agg1, hbf, N_NODES * FDIM, 1);
    k_gemm_wmma<<<gGemm, TB, 0, stream>>>(hbf, pWl2, bl2, xlb);
    k_gemm_wmma<<<gGemm, TB, 0, stream>>>(hbf, pWr2, br2, xrb);
    k_init_nodes<<<gElem, TB, 0, stream>>>(out, b2, smax, denom);
    k_edge_score<<<gWaveE, TB, 0, stream>>>(xlb, xrb, att2, src, dst, sbuf, smax);
    k_edge_exp<<<gEdge, TB, 0, stream>>>(sbuf, dst, smax, pbuf, denom);
    k_edge_agg<<<gWaveE, TB, 0, stream>>>(xlb, src, dst, pbuf, denom, out);
}
